// SpikeAttention_33105607917947
// MI455X (gfx1250) — matrix operand order repaired and checked by reading, not compiled, not run on hardware
//
#include <hip/hip_runtime.h>
#include <hip/hip_bf16.h>

typedef __attribute__((ext_vector_type(16))) _Float16 v16h;
typedef __attribute__((ext_vector_type(8)))  _Float16 v8h;
typedef __attribute__((ext_vector_type(4)))  _Float16 v4h;
typedef __attribute__((ext_vector_type(8)))  float    v8f;
typedef __attribute__((ext_vector_type(4)))  float    v4f;
typedef __attribute__((ext_vector_type(4)))  int      i32x4;

union V16 { v16h v; v8h h[2]; };

#define DMODEL 1024
#define NHEADS 16
#define HDIM   64
#define KV_G   64

#if defined(__HIP_DEVICE_COMPILE__) && \
    __has_builtin(__builtin_amdgcn_global_load_async_to_lds_b128) && \
    __has_builtin(__builtin_amdgcn_s_wait_asynccnt)
#define HAS_ASYNC_LDS 1
typedef __attribute__((address_space(1))) i32x4* as1_v4i;
typedef __attribute__((address_space(3))) i32x4* as3_v4i;
#define ASYNC_B128(gsrc, ldst) \
    __builtin_amdgcn_global_load_async_to_lds_b128( \
        (as1_v4i)(void*)(gsrc), (as3_v4i)(void*)(ldst), 0, 0)
#else
#define HAS_ASYNC_LDS 0
#endif

__device__ __forceinline__ float ms_act(float x) {
    float t = 4.0f * x;
    t = fminf(fmaxf(t, 0.0f), 4.0f);
    return floorf(t + 0.5f) * 0.25f;
}

// ---------------- elementwise: spike-quantize x -> f16 ----------------
__global__ void k_spike(const float* __restrict__ x, _Float16* __restrict__ xs, int total4) {
    int i = blockIdx.x * blockDim.x + threadIdx.x;
    if (i >= total4) return;
    v4f xv = *(const v4f*)(x + (size_t)i * 4);
    v4h r;
    r[0] = (_Float16)ms_act(xv[0]);
    r[1] = (_Float16)ms_act(xv[1]);
    r[2] = (_Float16)ms_act(xv[2]);
    r[3] = (_Float16)ms_act(xv[3]);
    *(v4h*)(xs + (size_t)i * 4) = r;
}

// ---------------- elementwise: f32 -> f16 weight convert ----------------
__global__ void k_cvt(const float* __restrict__ w, _Float16* __restrict__ w16, int total4) {
    int i = blockIdx.x * blockDim.x + threadIdx.x;
    if (i >= total4) return;
    v4f xv = *(const v4f*)(w + (size_t)i * 4);
    v4h r;
    r[0] = (_Float16)xv[0];
    r[1] = (_Float16)xv[1];
    r[2] = (_Float16)xv[2];
    r[3] = (_Float16)xv[3];
    *(v4h*)(w16 + (size_t)i * 4) = r;
}

__global__ void k_zero(float* __restrict__ p, int n) {
    int i = blockIdx.x * blockDim.x + threadIdx.x;
    if (i < n) p[i] = 0.0f;
}

// ---------------- big GEMM: out[M,C] = A[M,K] @ W[C,K]^T ----------------
// 256x128 block tile, 8 waves (4x2), each wave a 64x64 tile (4x4 WMMA tiles).
// Async path: double-buffered LDS, one barrier per K-step, DMA overlaps WMMA.
// apply_act=1: write ms_act(acc) as f16 into outH; else write acc f32 to outF.
__global__ __launch_bounds__(256)
void k_gemm(const _Float16* __restrict__ A, const _Float16* __restrict__ W,
            _Float16* __restrict__ outH, float* __restrict__ outF,
            int M, int K, int C, int apply_act) {
#if HAS_ASYNC_LDS
    __shared__ __align__(16) _Float16 sA[2 * 256 * 40];
    __shared__ __align__(16) _Float16 sB[2 * 128 * 40];
#else
    __shared__ __align__(16) _Float16 sA[256 * 40];
    __shared__ __align__(16) _Float16 sB[128 * 40];
#endif
    const int t    = threadIdx.x;
    const int lane = t & 31, w = t >> 5;
    const int wr = w >> 1, wc = w & 1;            // 4x2 wave grid
    const int colBase = blockIdx.x * 128;
    const int rowBase = blockIdx.y * 256;
    const int m0 = lane & 15, half = lane >> 4;
    const int nB = t >> 1, sgB = t & 1;

    v8f acc[4][4] = {};

    // per-thread compute over one LDS buffer (16 WMMAs)
    auto compute = [&](const _Float16* bA, const _Float16* bB) {
        V16 bfrag[4];
#pragma unroll
        for (int nb = 0; nb < 4; ++nb) {
            const _Float16* p = bB + (wc * 64 + nb * 16 + m0) * 40;
            bfrag[nb].h[0] = *(const v8h*)(p + half * 8);        // K = half*8 + e
            bfrag[nb].h[1] = *(const v8h*)(p + 16 + half * 8);
        }
#pragma unroll
        for (int mb = 0; mb < 4; ++mb) {
            V16 af;
            const _Float16* p = bA + (wr * 64 + mb * 16 + m0) * 40;
            af.h[0] = *(const v8h*)(p + half * 8);               // K = half*8 + e
            af.h[1] = *(const v8h*)(p + 16 + half * 8);          // K = 16 + half*8 + e
#pragma unroll
            for (int nb = 0; nb < 4; ++nb)
                acc[mb][nb] = __builtin_amdgcn_wmma_f32_16x16x32_f16(
                    false, af.v, false, bfrag[nb].v,
                    (short)0, acc[mb][nb], false, false);
        }
    };

#if HAS_ASYNC_LDS
    // issue this thread's portion of A/B tiles for K-offset kk into buffer buf
    auto stage = [&](int buf, int kk) {
        const _Float16* gA = A + (size_t)(rowBase + t) * K + kk;
        _Float16*       lA = sA + buf * (256 * 40) + t * 40;
        const _Float16* gB = W + (size_t)(colBase + nB) * K + kk + sgB * 16;
        _Float16*       lB = sB + buf * (128 * 40) + nB * 40 + sgB * 16;
        ASYNC_B128(gA,      lA);
        ASYNC_B128(gA + 8,  lA + 8);
        ASYNC_B128(gA + 16, lA + 16);
        ASYNC_B128(gA + 24, lA + 24);
        ASYNC_B128(gB,      lB);
        ASYNC_B128(gB + 8,  lB + 8);
    };
    const int nstep = K / 32;
    stage(0, 0);
    for (int s = 0; s < nstep; ++s) {
        const int cur = s & 1;
        __builtin_amdgcn_s_wait_asynccnt(0);  // this wave's loads into buf[cur] done
        __syncthreads();                      // all waves' loads done; all done reading buf[cur^1]
        if (s + 1 < nstep) stage(cur ^ 1, (s + 1) * 32);  // overlaps with compute
        compute(sA + cur * (256 * 40), sB + cur * (128 * 40));
    }
    __syncthreads();
#else
    for (int kk = 0; kk < K; kk += 32) {
        const _Float16* gA = A + (size_t)(rowBase + t) * K + kk;
        _Float16*       lA = sA + t * 40;
        const _Float16* gB = W + (size_t)(colBase + nB) * K + kk + sgB * 16;
        _Float16*       lB = sB + nB * 40 + sgB * 16;
        {
            const v8h* s = (const v8h*)gA;
            v8h a0 = s[0], a1 = s[1], a2 = s[2], a3 = s[3];
            *(v8h*)(lA)      = a0;
            *(v8h*)(lA + 8)  = a1;
            *(v8h*)(lA + 16) = a2;
            *(v8h*)(lA + 24) = a3;
            const v8h* sb = (const v8h*)gB;
            v8h b0 = sb[0], b1 = sb[1];
            *(v8h*)(lB)     = b0;
            *(v8h*)(lB + 8) = b1;
        }
        __syncthreads();
        compute(sA, sB);
        __syncthreads();
    }
#endif

#pragma unroll
    for (int mb = 0; mb < 4; ++mb)
#pragma unroll
        for (int nb = 0; nb < 4; ++nb) {
            int col = colBase + wc * 64 + nb * 16 + m0;
#pragma unroll
            for (int r = 0; r < 8; ++r) {
                int row = rowBase + wr * 64 + mb * 16 + r + 8 * half;
                float vres = acc[mb][nb][r];
                if (apply_act) outH[(size_t)row * C + col] = (_Float16)ms_act(vres);
                else           outF[(size_t)row * C + col] = vres;
            }
        }
}

// ---------------- per-head kv = k^T @ v (partial over N-chunk, atomic merge) ----------------
__global__ __launch_bounds__(256)
void k_kv_partial(const _Float16* __restrict__ kq, const _Float16* __restrict__ vq,
                  float* __restrict__ kv32, int N) {
    __shared__ __align__(16) _Float16 kT[64 * 40];
    __shared__ __align__(16) _Float16 vT[64 * 40];
    const int h = blockIdx.x;
    const int g = blockIdx.y;
    const int rowsPer = N / KV_G;
    const int r0 = g * rowsPer;
    const int t = threadIdx.x, lane = t & 31, w = t >> 5;
    const int m0 = lane & 15, half = lane >> 4;

    v8f acc[2] = {};

    for (int rr = 0; rr < rowsPer; rr += 32) {
        // cooperative load + transpose: kT[d][n], vT[e][n] for 32 rows
        {
            int row = t >> 3;            // 0..31
            int cs  = (t & 7) * 8;       // 0..56
            v8h kk8 = *(const v8h*)(kq + (size_t)(r0 + rr + row) * DMODEL + h * HDIM + cs);
            v8h vv8 = *(const v8h*)(vq + (size_t)(r0 + rr + row) * DMODEL + h * HDIM + cs);
#pragma unroll
            for (int j = 0; j < 8; ++j) {
                kT[(cs + j) * 40 + row] = kk8[j];
                vT[(cs + j) * 40 + row] = vv8[j];
            }
        }
        __syncthreads();
#pragma unroll
        for (int ti = 0; ti < 2; ++ti) {
            int tile = 2 * w + ti;       // 0..15 -> (mb, nb) in 4x4
            int mb = tile >> 2, nb = tile & 3;
            V16 af, bf;
            const _Float16* pa = kT + (mb * 16 + m0) * 40;
            af.h[0] = *(const v8h*)(pa + half * 8);
            af.h[1] = *(const v8h*)(pa + 16 + half * 8);
            const _Float16* pb = vT + (nb * 16 + m0) * 40;
            bf.h[0] = *(const v8h*)(pb + half * 8);
            bf.h[1] = *(const v8h*)(pb + 16 + half * 8);
            acc[ti] = __builtin_amdgcn_wmma_f32_16x16x32_f16(
                false, af.v, false, bf.v, (short)0, acc[ti], false, false);
        }
        __syncthreads();
    }

#pragma unroll
    for (int ti = 0; ti < 2; ++ti) {
        int tile = 2 * w + ti;
        int mb = tile >> 2, nb = tile & 3;
#pragma unroll
        for (int r = 0; r < 8; ++r) {
            int dM = mb * 16 + r + 8 * half;
            int eN = nb * 16 + m0;
            atomicAdd(&kv32[(h * HDIM + dM) * HDIM + eN], acc[ti][r]);
        }
    }
}

// ---------------- kv finalize: scale by 0.25, transpose, f16 ----------------
__global__ void k_kv_fin(const float* __restrict__ kv32, _Float16* __restrict__ kvT) {
    int i = blockIdx.x * blockDim.x + threadIdx.x;       // 16*64*64
    if (i >= NHEADS * HDIM * HDIM) return;
    int h = i >> 12, rem = i & 4095, d = rem >> 6, e = rem & 63;
    kvT[(h << 12) + (e << 6) + d] = (_Float16)(kv32[i] * 0.25f);
}

// ---------------- attn = q @ kv per head (K=64), fragments straight from global ----------------
__global__ __launch_bounds__(256)
void k_attn(const _Float16* __restrict__ q, const _Float16* __restrict__ kvT,
            _Float16* __restrict__ attn, int N) {
    const int t = threadIdx.x, lane = t & 31, w = t >> 5;
    const int m0 = lane & 15, half = lane >> 4;
    const int h  = blockIdx.y;
    const int r0 = blockIdx.x * 128 + w * 16;

    v8f acc[4] = {};
#pragma unroll
    for (int kb = 0; kb < 2; ++kb) {
        V16 af;
        const _Float16* pa = q + (size_t)(r0 + m0) * DMODEL + h * HDIM + kb * 32;
        af.h[0] = *(const v8h*)(pa + half * 8);
        af.h[1] = *(const v8h*)(pa + 16 + half * 8);
#pragma unroll
        for (int nb = 0; nb < 4; ++nb) {
            V16 bf;
            const _Float16* pb = kvT + ((size_t)h * HDIM + nb * 16 + m0) * HDIM + kb * 32;
            bf.h[0] = *(const v8h*)(pb + half * 8);
            bf.h[1] = *(const v8h*)(pb + 16 + half * 8);
            acc[nb] = __builtin_amdgcn_wmma_f32_16x16x32_f16(
                false, af.v, false, bf.v, (short)0, acc[nb], false, false);
        }
    }
#pragma unroll
    for (int nb = 0; nb < 4; ++nb) {
        int col = h * HDIM + nb * 16 + m0;
#pragma unroll
        for (int r = 0; r < 8; ++r) {
            int row = r0 + r + 8 * half;
            attn[(size_t)row * DMODEL + col] = (_Float16)acc[nb][r];
        }
    }
}

extern "C" void kernel_launch(void* const* d_in, const int* in_sizes, int n_in,
                              void* d_out, int out_size, void* d_ws, size_t ws_size,
                              hipStream_t stream) {
    const float* x  = (const float*)d_in[0];
    const float* Wq = (const float*)d_in[1];
    const float* Wk = (const float*)d_in[2];
    const float* Wv = (const float*)d_in[3];
    const float* Wp = (const float*)d_in[4];
    const int N = in_sizes[0] / DMODEL;
    float* out = (float*)d_out;

    char* ws = (char*)d_ws;
    const size_t SZH = (size_t)N * DMODEL * sizeof(_Float16);   // 128 MB
    const size_t WSZ = (size_t)DMODEL * DMODEL;                 // elements per weight
    _Float16* xs   = (_Float16*)(ws);              // also reused for attn
    _Float16* qb   = (_Float16*)(ws + SZH);
    _Float16* kb   = (_Float16*)(ws + 2 * SZH);
    _Float16* vb   = (_Float16*)(ws + 3 * SZH);
    _Float16* w16  = (_Float16*)(ws + 4 * SZH);    // 4 x D*D f16
    float*    kv32 = (float*)(ws + 4 * SZH + 4 * WSZ * sizeof(_Float16));
    _Float16* kvT  = (_Float16*)((char*)kv32 + (size_t)NHEADS * HDIM * HDIM * sizeof(float));

    // 1. spike-quantize x
    {
        int total4 = (N * DMODEL) / 4;
        k_spike<<<(total4 + 255) / 256, 256, 0, stream>>>(x, xs, total4);
    }
    // 2. weights -> f16
    {
        int total4 = (DMODEL * DMODEL) / 4;
        int gb = (total4 + 255) / 256;
        k_cvt<<<gb, 256, 0, stream>>>(Wq, w16 + 0 * WSZ, total4);
        k_cvt<<<gb, 256, 0, stream>>>(Wk, w16 + 1 * WSZ, total4);
        k_cvt<<<gb, 256, 0, stream>>>(Wv, w16 + 2 * WSZ, total4);
        k_cvt<<<gb, 256, 0, stream>>>(Wp, w16 + 3 * WSZ, total4);
    }
    // 3. q,k,v GEMMs with fused ms_act
    dim3 gg(DMODEL / 128, N / 256);
    k_gemm<<<gg, 256, 0, stream>>>(xs, w16 + 0 * WSZ, qb, nullptr, N, DMODEL, DMODEL, 1);
    k_gemm<<<gg, 256, 0, stream>>>(xs, w16 + 1 * WSZ, kb, nullptr, N, DMODEL, DMODEL, 1);
    k_gemm<<<gg, 256, 0, stream>>>(xs, w16 + 2 * WSZ, vb, nullptr, N, DMODEL, DMODEL, 1);
    // 4. kv = k^T v per head
    k_zero<<<(NHEADS * HDIM * HDIM + 255) / 256, 256, 0, stream>>>(kv32, NHEADS * HDIM * HDIM);
    dim3 gkv(NHEADS, KV_G);
    k_kv_partial<<<gkv, 256, 0, stream>>>(kb, vb, kv32, N);
    k_kv_fin<<<(NHEADS * HDIM * HDIM + 255) / 256, 256, 0, stream>>>(kv32, kvT);
    // 5. attn = q @ (kv * SCALE)  (scale folded into kvT)
    dim3 ga(N / 128, NHEADS);
    k_attn<<<ga, 256, 0, stream>>>(qb, kvT, xs /* attn reuses xs buffer */, N);
    // 6. out = attn @ Wp^T (f32)
    k_gemm<<<gg, 256, 0, stream>>>(xs, w16 + 3 * WSZ, nullptr, out, N, DMODEL, DMODEL, 0);
}
